// GraphSAGE_81423989997900
// MI455X (gfx1250) — matrix operand order repaired and checked by reading, not compiled, not run on hardware
//
#include <hip/hip_runtime.h>
#include <hip/hip_bf16.h>

#define D_IN  128
#define D_H   128
#define D_OUT 64

typedef __attribute__((ext_vector_type(16))) __bf16 v16bf;
typedef __attribute__((ext_vector_type(8)))  float  v8f;

// ---------------------------------------------------------------------------
// A-operand loader: 16x32 bf16 tile, per-lane slice.
// lanes 0-15  (row M=lane):    K = kk+{0..7}  then kk+16+{0..7}
// lanes 16-31 (row M=lane-16): K = kk+8+{0..7} then kk+24+{0..7}
// caller passes p = &A[row*K + kk + kGroup*8]; second half at p+16.
// scale is applied before bf16 conversion (fused mean normalization).
// ---------------------------------------------------------------------------
__device__ __forceinline__ v16bf load_A_k32(const float* __restrict__ p, float s) {
    float4 a0 = *(const float4*)(p);
    float4 a1 = *(const float4*)(p + 4);
    float4 b0 = *(const float4*)(p + 16);
    float4 b1 = *(const float4*)(p + 20);
    v16bf r;
    r[0]  = (__bf16)(a0.x * s); r[1]  = (__bf16)(a0.y * s);
    r[2]  = (__bf16)(a0.z * s); r[3]  = (__bf16)(a0.w * s);
    r[4]  = (__bf16)(a1.x * s); r[5]  = (__bf16)(a1.y * s);
    r[6]  = (__bf16)(a1.z * s); r[7]  = (__bf16)(a1.w * s);
    r[8]  = (__bf16)(b0.x * s); r[9]  = (__bf16)(b0.y * s);
    r[10] = (__bf16)(b0.z * s); r[11] = (__bf16)(b0.w * s);
    r[12] = (__bf16)(b1.x * s); r[13] = (__bf16)(b1.y * s);
    r[14] = (__bf16)(b1.z * s); r[15] = (__bf16)(b1.w * s);
    return r;
}

// ---------------------------------------------------------------------------
// B-operand loader: 32x16 bf16 tile (B = W^T, column n holds W[n, :] along K).
// lanes 0-15  (col N=lane):    K = kk+{0..7}  then kk+16+{0..7}
// lanes 16-31 (col N=lane-16): K = kk+8+{0..7} then kk+24+{0..7}
// caller passes p = &W[col*K + kk + kGroup*8]; second half at p+16.
// ---------------------------------------------------------------------------
__device__ __forceinline__ v16bf load_B_k32(const float* __restrict__ p) {
    float4 a0 = *(const float4*)(p);
    float4 a1 = *(const float4*)(p + 4);
    float4 a2 = *(const float4*)(p + 16);
    float4 a3 = *(const float4*)(p + 20);
    v16bf r;
    r[0]  = (__bf16)a0.x; r[1]  = (__bf16)a0.y; r[2]  = (__bf16)a0.z; r[3]  = (__bf16)a0.w;
    r[4]  = (__bf16)a1.x; r[5]  = (__bf16)a1.y; r[6]  = (__bf16)a1.z; r[7]  = (__bf16)a1.w;
    r[8]  = (__bf16)a2.x; r[9]  = (__bf16)a2.y; r[10] = (__bf16)a2.z; r[11] = (__bf16)a2.w;
    r[12] = (__bf16)a3.x; r[13] = (__bf16)a3.y; r[14] = (__bf16)a3.z; r[15] = (__bf16)a3.w;
    return r;
}

// ---------------------------------------------------------------------------
// Fused (dual-)GEMM:  out = [relu]( A1n @ W1^T + bias (+ A2 @ W2^T) )
// where A1n = A1 * 1/max(cnt,1) per row (cnt==null -> identity).
// One wave computes one 16x16 output tile via v_wmma_f32_16x16x32_bf16.
// grid.x = nRows/16, blockDim.x = 2*Mout (i.e. Mout/16 waves per block).
// nRows % 16 == 0 -> no predication, EXEC all-ones at every WMMA.
// ---------------------------------------------------------------------------
__global__ void sage_gemm_wmma(const float* __restrict__ A1,
                               const float* __restrict__ W1,
                               const float* __restrict__ bias,
                               const float* __restrict__ cnt,
                               const float* __restrict__ A2,
                               const float* __restrict__ W2,
                               float* __restrict__ out,
                               int K, int Mout, int doRelu) {
    const int lane    = threadIdx.x & 31;
    const int wave    = threadIdx.x >> 5;
    const int rowBase = blockIdx.x * 16;
    const int colBase = wave * 16;

    const int l15   = lane & 15;
    const int half  = lane >> 4;          // 0 or 1
    const int aRow  = rowBase + l15;      // this lane's A row
    const int bCol  = colBase + l15;      // this lane's W row (= output col)

    float scale1 = 1.0f;
    if (cnt) scale1 = 1.0f / fmaxf(cnt[aRow], 1.0f);

    const float* a1p = A1 + (size_t)aRow * K + half * 8;   // + kk, +16 for 2nd octet
    const float* b1p = W1 + (size_t)bCol * K + half * 8;   // + kk, +16 for 2nd octet

    v8f acc = {};
    #pragma unroll 4
    for (int kk = 0; kk < K; kk += 32) {
        v16bf a = load_A_k32(a1p + kk, scale1);
        v16bf b = load_B_k32(b1p + kk);
        acc = __builtin_amdgcn_wmma_f32_16x16x32_bf16(
                  false, a, false, b, (short)0, acc, false, false);
    }
    if (A2) {
        const float* a2p = A2 + (size_t)aRow * K + half * 8;
        const float* b2p = W2 + (size_t)bCol * K + half * 8;
        #pragma unroll 4
        for (int kk = 0; kk < K; kk += 32) {
            v16bf a = load_A_k32(a2p + kk, 1.0f);
            v16bf b = load_B_k32(b2p + kk);
            acc = __builtin_amdgcn_wmma_f32_16x16x32_bf16(
                      false, a, false, b, (short)0, acc, false, false);
        }
    }

    // D layout: VGPR r, lanes 0-15 -> (M=r, N=lane); lanes 16-31 -> (M=r+8, N=lane-16)
    const int   n  = colBase + l15;
    const float bv = bias ? bias[n] : 0.0f;
    const int   m0 = rowBase + half * 8;
    #pragma unroll
    for (int r = 0; r < 8; ++r) {
        float v = acc[r] + bv;
        if (doRelu) v = fmaxf(v, 0.0f);
        out[(size_t)(m0 + r) * Mout + n] = v;
    }
}

// ---------------------------------------------------------------------------
// In-degree counts (float, used as divisor for mean aggregation).
// ---------------------------------------------------------------------------
__global__ void count_edges(const long long* __restrict__ dst,
                            float* __restrict__ cnt, int E) {
    int e = blockIdx.x * blockDim.x + threadIdx.x;
    if (e < E) atomicAdd(&cnt[dst[e]], 1.0f);
}

// ---------------------------------------------------------------------------
// Edge scatter: one wave per edge; lane-stride-32 so the gather from h[src]
// and the 128 atomic f32 adds into agg[dst] are both fully coalesced.
// h/agg (51 MB each) are L2-resident on MI455X (192 MB L2).
// ---------------------------------------------------------------------------
__global__ void scatter_edges(const long long* __restrict__ src,
                              const long long* __restrict__ dst,
                              const float* __restrict__ h,
                              float* __restrict__ agg, int E) {
    const int lane = threadIdx.x & 31;
    const int e    = blockIdx.x * (blockDim.x >> 5) + (threadIdx.x >> 5);
    if (e >= E) return;
    const long long s = src[e];
    const long long d = dst[e];
    const float* hp = h   + (size_t)s * D_H;
    float*       ap = agg + (size_t)d * D_H;
    #pragma unroll
    for (int j = 0; j < D_H / 32; ++j) {
        const int f = lane + 32 * j;
        atomicAdd(ap + f, hp[f]);
    }
}

// ---------------------------------------------------------------------------
extern "C" void kernel_launch(void* const* d_in, const int* in_sizes, int n_in,
                              void* d_out, int out_size, void* d_ws, size_t ws_size,
                              hipStream_t stream) {
    const float*     x     = (const float*)d_in[0];
    const long long* edge  = (const long long*)d_in[1];   // int64 [2, E]
    const float*     emb_W = (const float*)d_in[2];
    const float*     emb_b = (const float*)d_in[3];
    const float*     Wl0   = (const float*)d_in[4];
    const float*     bl0   = (const float*)d_in[5];
    const float*     Wr0   = (const float*)d_in[6];
    const float*     Wl1   = (const float*)d_in[7];
    const float*     bl1   = (const float*)d_in[8];
    const float*     Wr1   = (const float*)d_in[9];
    const float*     Wl2   = (const float*)d_in[10];
    const float*     bl2   = (const float*)d_in[11];
    const float*     Wr2   = (const float*)d_in[12];

    const int N = in_sizes[0] / D_IN;   // 100000 (divisible by 16)
    const int E = in_sizes[1] / 2;      // 1600000

    const long long* srcIdx = edge;
    const long long* dstIdx = edge + E;

    // workspace: hA | hB | agg | cnt   (~154 MB)
    float* hA  = (float*)d_ws;
    float* hB  = hA  + (size_t)N * D_H;
    float* agg = hB  + (size_t)N * D_H;
    float* cnt = agg + (size_t)N * D_H;

    const dim3 gGemm(N / 16);
    const dim3 bGemmH(2 * D_H);    // 8 waves: 16 rows x 128 cols per block
    const dim3 bGemmO(2 * D_OUT);  // 4 waves: 16 rows x  64 cols per block
    const dim3 gCnt((E + 255) / 256), bCnt(256);
    const dim3 gSc((E + 7) / 8),      bSc(256);   // 8 edges (waves) per block

    // degree counts (edge structure is layer-invariant)
    hipMemsetAsync(cnt, 0, (size_t)N * sizeof(float), stream);
    count_edges<<<gCnt, bCnt, 0, stream>>>(dstIdx, cnt, E);

    // embedding: hA = x @ emb_W^T + emb_b
    sage_gemm_wmma<<<gGemm, bGemmH, 0, stream>>>(
        x, emb_W, emb_b, nullptr, nullptr, nullptr, hA, D_IN, D_H, 0);

    // layer 0: hB = relu(mean(hA) @ Wl0^T + bl0 + hA @ Wr0^T)
    hipMemsetAsync(agg, 0, (size_t)N * D_H * sizeof(float), stream);
    scatter_edges<<<gSc, bSc, 0, stream>>>(srcIdx, dstIdx, hA, agg, E);
    sage_gemm_wmma<<<gGemm, bGemmH, 0, stream>>>(
        agg, Wl0, bl0, cnt, hA, Wr0, hB, D_H, D_H, 1);

    // layer 1: hA = relu(mean(hB) @ Wl1^T + bl1 + hB @ Wr1^T)
    hipMemsetAsync(agg, 0, (size_t)N * D_H * sizeof(float), stream);
    scatter_edges<<<gSc, bSc, 0, stream>>>(srcIdx, dstIdx, hB, agg, E);
    sage_gemm_wmma<<<gGemm, bGemmH, 0, stream>>>(
        agg, Wl1, bl1, cnt, hB, Wr1, hA, D_H, D_H, 1);

    // layer 2: out = mean(hA) @ Wl2^T + bl2 + hA @ Wr2^T   (no relu)
    hipMemsetAsync(agg, 0, (size_t)N * D_H * sizeof(float), stream);
    scatter_edges<<<gSc, bSc, 0, stream>>>(srcIdx, dstIdx, hA, agg, E);
    sage_gemm_wmma<<<gGemm, bGemmO, 0, stream>>>(
        agg, Wl2, bl2, cnt, hA, Wr2, (float*)d_out, D_H, D_OUT, 0);
}
